// TrackedAttention_12137577578828
// MI455X (gfx1250) — matrix operand order repaired and checked by reading, not compiled, not run on hardware
//
#include <hip/hip_runtime.h>
#include <hip/hip_bf16.h>

typedef __attribute__((ext_vector_type(16))) _Float16 v16h;
typedef __attribute__((ext_vector_type(8)))  _Float16 v8h;
typedef __attribute__((ext_vector_type(8)))  float    v8f;

#define B_  2
#define S_  2048
#define D_  1024
#define H_  16
#define HD_ 64
#define M_TOK (B_*S_)   // 4096
#define LPAD 72         // padded LDS row (halfs): 144B, bank-conflict-free row stride

union V16H { v16h v; v8h h[2]; };
union V8F  { v8f v; float f[8]; };

__device__ __forceinline__ float rowmax16(float v) {
  v = fmaxf(v, __shfl_xor(v, 1, 32));
  v = fmaxf(v, __shfl_xor(v, 2, 32));
  v = fmaxf(v, __shfl_xor(v, 4, 32));
  v = fmaxf(v, __shfl_xor(v, 8, 32));
  return v;
}
__device__ __forceinline__ float rowsum16(float v) {
  v += __shfl_xor(v, 1, 32);
  v += __shfl_xor(v, 2, 32);
  v += __shfl_xor(v, 4, 32);
  v += __shfl_xor(v, 8, 32);
  return v;
}

// ---- CDNA5 async global->LDS copy (ASYNCcnt-tracked), inline asm ----
__device__ __forceinline__ void async_copy_b128(const _Float16* lds_dst, const _Float16* gsrc) {
  // LDS-aperture flat address truncates to the LDS byte address (ISA 10.2)
  uint32_t lds_off = (uint32_t)(uintptr_t)lds_dst;
  asm volatile("global_load_async_to_lds_b128 %0, %1, off"
               :: "v"(lds_off), "v"(gsrc) : "memory");
}
template <int N>
__device__ __forceinline__ void wait_asynccnt() {
  asm volatile("s_wait_asynccnt %0" :: "i"(N) : "memory");
}

// ---------------- f32 -> f16 convert ----------------
__global__ void cvt_f32_to_f16(const float* __restrict__ x, _Float16* __restrict__ y, int n) {
  int i = blockIdx.x * blockDim.x + threadIdx.x;
  if (i < n) y[i] = (_Float16)x[i];
}

// ---------------- transpose weights to n-major f16: Wt[z][n][k] = W[z][k][n] ----
__global__ void transpose_w(const float* __restrict__ Wq, const float* __restrict__ Wk,
                            const float* __restrict__ Wv, _Float16* __restrict__ Wt) {
  const float* W = (blockIdx.y == 0) ? Wq : ((blockIdx.y == 1) ? Wk : Wv);
  _Float16* out = Wt + (size_t)blockIdx.y * D_ * D_;
  int i = blockIdx.x * blockDim.x + threadIdx.x;
  int n = i / D_, k = i % D_;
  out[i] = (_Float16)W[k * D_ + n];
}

// ---------------- QKV projection: Y = X*W + b, WMMA f16 ----------------
__global__ void __launch_bounds__(128)
qkv_gemm(const _Float16* __restrict__ Xh,
         const _Float16* __restrict__ Wt,
         const float* __restrict__ bq, const float* __restrict__ bk,
         const float* __restrict__ bv,
         _Float16* __restrict__ Qh, _Float16* __restrict__ Kh,
         _Float16* __restrict__ Vt) {
  const int z    = blockIdx.z;
  const int wave = threadIdx.x >> 5, lane = threadIdx.x & 31;
  const int hl   = lane >> 4, lr = lane & 15;
  const int m_base = blockIdx.y * 64 + wave * 16;
  const int n_base = blockIdx.x * 64;
  const _Float16* w    = Wt + (size_t)z * D_ * D_;
  const float*    bias = (z == 0) ? bq : ((z == 1) ? bk : bv);

  const _Float16* arow = Xh + (size_t)(m_base + lr) * D_;

  V8F acc[4];
  #pragma unroll
  for (int t = 0; t < 4; t++)
    #pragma unroll
    for (int r = 0; r < 8; r++) acc[t].f[r] = 0.f;

  for (int k0 = 0; k0 < D_; k0 += 32) {
    V16H a;
    a.h[0] = *(const v8h*)(arow + k0 +      hl * 16);
    a.h[1] = *(const v8h*)(arow + k0 + 8 + hl * 16);
    v16h bf[4];
    #pragma unroll
    for (int t = 0; t < 4; t++)
      bf[t] = *(const v16h*)(w + (size_t)(n_base + t * 16 + lr) * D_ + k0 + hl * 16);
    #pragma unroll
    for (int t = 0; t < 4; t++)
      acc[t].v = __builtin_amdgcn_wmma_f32_16x16x32_f16(
          false, a.v, false, bf[t], (short)0, acc[t].v, false, false);
  }

  #pragma unroll
  for (int t = 0; t < 4; t++) {
    int n = n_base + t * 16 + lr;
    float bval = bias[n];
    int hh = n >> 6, hd = n & 63;
    #pragma unroll
    for (int r = 0; r < 8; r++) {
      int m = m_base + r + 8 * hl;
      int b = m >> 11, s = m & 2047;
      _Float16 val = (_Float16)(acc[t].f[r] + bval);
      if (z == 0)      Qh[(((size_t)(b * H_ + hh)) * S_ + s) * HD_ + hd] = val;
      else if (z == 1) Kh[(((size_t)(b * H_ + hh)) * S_ + s) * HD_ + hd] = val;
      else             Vt[(((size_t)(b * H_ + hh)) * HD_ + hd) * S_ + s] = val;
    }
  }
}

// ---------------- fused flash attention with async double-buffered LDS ----------------
__global__ void __launch_bounds__(128)
flash_attn(const _Float16* __restrict__ Qh, const _Float16* __restrict__ Kh,
           const _Float16* __restrict__ Vt, const float* __restrict__ mask,
           float* __restrict__ out) {
  const int b = blockIdx.z, h = blockIdx.y, qt = blockIdx.x;
  const int tid = threadIdx.x;
  const int wave = tid >> 5, lane = tid & 31;
  const int hl = lane >> 4, lr = lane & 15;
  const int q_base = qt * 64 + wave * 16;

  __shared__ __align__(32) _Float16 Klds[2][64][LPAD];   // key-major
  __shared__ __align__(32) _Float16 Vlds[2][64][LPAD];   // hd-major
  __shared__ __align__(32) _Float16 Plds[4][16][LPAD];

  const size_t bh = (size_t)(b * H_ + h);
  const _Float16* Kbase = Kh + (bh * S_) * HD_;          // [S][64] contiguous
  const _Float16* Vbase = Vt + (bh * HD_) * S_;          // [64][S] rows strided S_

  // ---- async stage of one 64-key chunk: 8 b128 per thread ----
  auto stage_chunk = [&](int kv0, int buf) {
    #pragma unroll
    for (int k = 0; k < 4; k++) {
      int seg = tid + k * 128;              // 0..511
      int row = seg >> 3, c8 = (seg & 7) * 8;
      async_copy_b128(&Klds[buf][row][c8], Kbase + (size_t)(kv0) * HD_ + seg * 8);
      async_copy_b128(&Vlds[buf][row][c8], Vbase + (size_t)row * S_ + kv0 + c8);
    }
  };

  // Q A-fragments, persistent
  const _Float16* qrow = Qh + (bh * S_ + q_base + lr) * HD_;
  V16H aq0, aq1;
  aq0.h[0] = *(const v8h*)(qrow +      hl * 16);
  aq0.h[1] = *(const v8h*)(qrow + 8 + hl * 16);
  aq1.h[0] = *(const v8h*)(qrow + 32 + hl * 16);
  aq1.h[1] = *(const v8h*)(qrow + 40 + hl * 16);

  V8F o[4];
  float m_i[8], l_i[8];
  #pragma unroll
  for (int r = 0; r < 8; r++) { m_i[r] = -1e30f; l_i[r] = 0.f; }
  #pragma unroll
  for (int t = 0; t < 4; t++)
    #pragma unroll
    for (int r = 0; r < 8; r++) o[t].f[r] = 0.f;

  const float* maskb = mask + (size_t)b * S_;
  const int NCH = S_ / 64;

  stage_chunk(0, 0);

  for (int c = 0; c < NCH; c++) {
    const int buf = c & 1;
    if (c + 1 < NCH) {
      stage_chunk((c + 1) * 64, buf ^ 1);   // prefetch next chunk
      wait_asynccnt<8>();                   // in-order: chunk c's 8 are done
    } else {
      wait_asynccnt<0>();
    }
    __syncthreads();

    // ---- scores S = (Q K^T)/8 + mask, B-frags from LDS ----
    V8F sj[4];
    #pragma unroll
    for (int j = 0; j < 4; j++) {
      const _Float16* krow = &Klds[buf][j * 16 + lr][0];
      v16h kb0 = *(const v16h*)(krow +      hl * 16);
      v16h kb1 = *(const v16h*)(krow + 32 + hl * 16);
      V8F cz;
      #pragma unroll
      for (int r = 0; r < 8; r++) cz.f[r] = 0.f;
      v8f s = __builtin_amdgcn_wmma_f32_16x16x32_f16(
          false, aq0.v, false, kb0, (short)0, cz.v, false, false);
      s = __builtin_amdgcn_wmma_f32_16x16x32_f16(
          false, aq1.v, false, kb1, (short)0, s, false, false);
      float mk = maskb[c * 64 + j * 16 + lr];
      sj[j].v = s;
      #pragma unroll
      for (int r = 0; r < 8; r++) sj[j].f[r] = sj[j].f[r] * 0.125f + mk;
    }

    // ---- online softmax ----
    #pragma unroll
    for (int r = 0; r < 8; r++) {
      float mx = fmaxf(fmaxf(sj[0].f[r], sj[1].f[r]), fmaxf(sj[2].f[r], sj[3].f[r]));
      mx = rowmax16(mx);
      float mnew = fmaxf(m_i[r], mx);
      float corr = __expf(m_i[r] - mnew);
      m_i[r] = mnew;
      float psum = 0.f;
      #pragma unroll
      for (int j = 0; j < 4; j++) {
        float p = __expf(sj[j].f[r] - mnew);
        sj[j].f[r] = p;
        psum += p;
      }
      psum = rowsum16(psum);
      l_i[r] = l_i[r] * corr + psum;
      #pragma unroll
      for (int t = 0; t < 4; t++) o[t].f[r] *= corr;
    }

    // ---- stage P (f16) in LDS, re-read as A-fragments ----
    #pragma unroll
    for (int j = 0; j < 4; j++)
      #pragma unroll
      for (int r = 0; r < 8; r++)
        Plds[wave][r + 8 * hl][j * 16 + lr] = (_Float16)sj[j].f[r];

    V16H ap0, ap1;
    ap0.h[0] = *(const v8h*)(&Plds[wave][lr][     hl * 16]);
    ap0.h[1] = *(const v8h*)(&Plds[wave][lr][8 + hl * 16]);
    ap1.h[0] = *(const v8h*)(&Plds[wave][lr][32 + hl * 16]);
    ap1.h[1] = *(const v8h*)(&Plds[wave][lr][40 + hl * 16]);

    // ---- O += P * V, V B-frags from LDS ----
    #pragma unroll
    for (int t = 0; t < 4; t++) {
      const _Float16* vrow = &Vlds[buf][t * 16 + lr][0];
      v16h vb0 = *(const v16h*)(vrow +      hl * 16);
      v16h vb1 = *(const v16h*)(vrow + 32 + hl * 16);
      o[t].v = __builtin_amdgcn_wmma_f32_16x16x32_f16(
          false, ap0.v, false, vb0, (short)0, o[t].v, false, false);
      o[t].v = __builtin_amdgcn_wmma_f32_16x16x32_f16(
          false, ap1.v, false, vb1, (short)0, o[t].v, false, false);
    }
    __syncthreads();   // all reads of buf done before it is overwritten next iter
  }

  // ---- normalize and write ctx [B,S,D] f32 ----
  #pragma unroll
  for (int r = 0; r < 8; r++) {
    float inv = 1.f / l_i[r];
    int srow = q_base + r + 8 * hl;
    float* orow = out + ((size_t)b * S_ + srow) * D_ + h * HD_;
    #pragma unroll
    for (int t = 0; t < 4; t++)
      orow[t * 16 + lr] = o[t].f[r] * inv;
  }
}

extern "C" void kernel_launch(void* const* d_in, const int* in_sizes, int n_in,
                              void* d_out, int out_size, void* d_ws, size_t ws_size,
                              hipStream_t stream) {
  (void)in_sizes; (void)n_in; (void)out_size; (void)ws_size;
  const float* hs   = (const float*)d_in[0];
  const float* mask = (const float*)d_in[1];
  const float* Wq   = (const float*)d_in[2];
  const float* bq   = (const float*)d_in[3];
  const float* Wk   = (const float*)d_in[4];
  const float* bk   = (const float*)d_in[5];
  const float* Wv   = (const float*)d_in[6];
  const float* bv   = (const float*)d_in[7];
  float* out = (float*)d_out;

  char* ws = (char*)d_ws;
  _Float16* Xh = (_Float16*)(ws);                            //  8 MB
  _Float16* Wt = (_Float16*)(ws + (size_t) 8 * 1024 * 1024); //  6 MB
  _Float16* Qh = (_Float16*)(ws + (size_t)14 * 1024 * 1024); //  8 MB [B,H,S,HD]
  _Float16* Kh = (_Float16*)(ws + (size_t)22 * 1024 * 1024); //  8 MB [B,H,S,HD]
  _Float16* Vt = (_Float16*)(ws + (size_t)30 * 1024 * 1024); //  8 MB [B,H,HD,S]

  cvt_f32_to_f16<<<dim3((M_TOK * D_) / 256), dim3(256), 0, stream>>>(hs, Xh, M_TOK * D_);
  transpose_w<<<dim3((D_ * D_) / 256, 3), dim3(256), 0, stream>>>(Wq, Wk, Wv, Wt);
  qkv_gemm<<<dim3(D_ / 64, M_TOK / 64, 3), dim3(128), 0, stream>>>(Xh, Wt, bq, bk, bv, Qh, Kh, Vt);
  flash_attn<<<dim3(S_ / 64, H_, B_), dim3(128), 0, stream>>>(Qh, Kh, Vt, mask, out);
}
